// TripleViewNet_32263794327573
// MI455X (gfx1250) — matrix operand order repaired and checked by reading, not compiled, not run on hardware
//
#include <hip/hip_runtime.h>
#include <hip/hip_bf16.h>
#include <math.h>

#define NTOT   32768
#define FDIM   768
#define HIDD   128
#define NHEADS 4
#define HDIM   32
#define BGR    64
#define NPG    512
#define NEDGE  262144
#define ETOT   (NEDGE + NTOT)
#define KPOOL  256
#define EPSV   1e-5f

typedef __bf16 bf16;
typedef __attribute__((ext_vector_type(16))) __bf16 v16bf;
typedef __attribute__((ext_vector_type(8)))  __bf16 v8bf;
typedef __attribute__((ext_vector_type(8)))  float  v8f;

// ---------- helpers ----------
__device__ __forceinline__ unsigned fenc(float x) {
  int i = __float_as_int(x);
  return (i >= 0) ? ((unsigned)i | 0x80000000u) : ~((unsigned)i);
}
__device__ __forceinline__ float fdec(unsigned k) {
  int i = (k & 0x80000000u) ? (int)(k & 0x7fffffffu) : (int)(~k);
  return __int_as_float(i);
}

// ---------- fill ----------
__global__ void fill_u32_kernel(unsigned* __restrict__ p, unsigned v, int n) {
  int i = blockIdx.x * blockDim.x + threadIdx.x;
  if (i < n) p[i] = v;
}

// ---------- per-row LayerNorm stats (wave per row) ----------
__global__ __launch_bounds__(256)
void row_stats_kernel(const float* __restrict__ x, float* __restrict__ mu, float* __restrict__ rstd) {
  int row  = blockIdx.x * 8 + (threadIdx.x >> 5);
  int lane = threadIdx.x & 31;
  const float* xr = x + (size_t)row * FDIM;
  float s = 0.f, s2 = 0.f;
  for (int i = lane; i < FDIM; i += 32) { float v = xr[i]; s += v; s2 += v * v; }
  for (int o = 16; o > 0; o >>= 1) { s += __shfl_xor(s, o); s2 += __shfl_xor(s2, o); }
  if (lane == 0) {
    float m = s * (1.f / FDIM);
    float var = s2 * (1.f / FDIM) - m * m;
    mu[row] = m; rstd[row] = rsqrtf(var + EPSV);
  }
}

// ---------- pack f32 KxN weight into WMMA-native bf16 B layout ----------
// element (kt,nt,lane,e): K = 32*kt + 8*(lane>=16) + (e<8 ? e : 8+e), N = 16*nt + lane%16
__global__ void pack_b_kernel(const float* __restrict__ W, bf16* __restrict__ out, int Kd, int N) {
  int idx = blockIdx.x * blockDim.x + threadIdx.x;
  if (idx >= Kd * N) return;
  int e = idx & 15, lane = (idx >> 4) & 31, tile = idx >> 9;
  int ntiles = N >> 4;
  int kt = tile / ntiles, nt = tile % ntiles;
  int kk = kt * 32 + 8 * (lane >> 4) + (e < 8 ? e : 8 + e);
  int nn = nt * 16 + (lane & 15);
  out[idx] = (bf16)W[(size_t)kk * N + nn];
}

// ---------- WMMA GEMM ----------
// mode 0: A = f32 x with fused LayerNorm; epilogue bias+leaky(0.01) -> bf16 Cbf
// mode 1: A = bf16; epilogue raw f32 -> Cf
__global__ __launch_bounds__(128)
void gemm_wmma_kernel(const void* __restrict__ Asrc, const bf16* __restrict__ Bpack,
                      float* __restrict__ Cf, bf16* __restrict__ Cbf,
                      const float* __restrict__ mu, const float* __restrict__ rstd,
                      const float* __restrict__ lng, const float* __restrict__ lnb,
                      const float* __restrict__ bias, int N, int Kd, int mode) {
  const int wave = threadIdx.x >> 5;
  const int lane = threadIdx.x & 31;
  const int lm = lane & 15, hi = lane >> 4;
  const int rowBase = blockIdx.x * 64 + wave * 16;
  const int colBase = blockIdx.y * 128;
  const int m = rowBase + lm;
  v8f acc[8] = {};

  for (int k = 0; k < Kd; k += 32) {
    v16bf a;
    if (mode == 0) {
      const float* xr = (const float*)Asrc + (size_t)m * Kd;
      const float muv = mu[m], rs = rstd[m];
      const int c0 = k + 8 * hi, c1 = k + 16 + 8 * hi;
#pragma unroll
      for (int e = 0; e < 8; ++e) {
        a[e]     = (bf16)((xr[c0 + e] - muv) * rs * lng[c0 + e] + lnb[c0 + e]);
        a[e + 8] = (bf16)((xr[c1 + e] - muv) * rs * lng[c1 + e] + lnb[c1 + e]);
      }
    } else {
      const bf16* ar = (const bf16*)Asrc + (size_t)m * Kd;
      v8bf lo = *(const v8bf*)(ar + k + 8 * hi);
      v8bf hc = *(const v8bf*)(ar + k + 16 + 8 * hi);
#pragma unroll
      for (int e = 0; e < 8; ++e) { a[e] = lo[e]; a[e + 8] = hc[e]; }
    }
    const int kt = k >> 5;
    const bf16* bbase = Bpack + (((size_t)kt * (N >> 4) + (colBase >> 4)) * 32 + lane) * 16;
#pragma unroll
    for (int j = 0; j < 8; ++j) {
      v16bf b = *(const v16bf*)(bbase + (size_t)j * 512);
      acc[j] = __builtin_amdgcn_wmma_f32_16x16x32_bf16(false, a, false, b, (short)0, acc[j], false, false);
    }
  }
#pragma unroll
  for (int j = 0; j < 8; ++j) {
    const int col = colBase + j * 16 + lm;
#pragma unroll
    for (int r = 0; r < 8; ++r) {
      const int row = rowBase + r + 8 * hi;
      float v = acc[j][r];
      if (mode == 0) {
        v += bias[col];
        v = v > 0.f ? v : 0.01f * v;
        Cbf[(size_t)row * N + col] = (bf16)v;
      } else {
        Cf[(size_t)row * N + col] = v;
      }
    }
  }
}

// ---------- GAT attention coefficients ----------
__global__ void attn_coeff_kernel(const float* __restrict__ hh, const float* __restrict__ asrc,
                                  const float* __restrict__ adst, float* __restrict__ a_s,
                                  float* __restrict__ a_d) {
  int idx = blockIdx.x * blockDim.x + threadIdx.x;
  if (idx >= NTOT * NHEADS) return;
  int n = idx >> 2, h = idx & 3;
  const float* hr = hh + (size_t)n * HIDD + h * HDIM;
  float s = 0.f, d = 0.f;
#pragma unroll
  for (int c = 0; c < HDIM; ++c) { s += hr[c] * asrc[h * HDIM + c]; d += hr[c] * adst[h * HDIM + c]; }
  a_s[idx] = s; a_d[idx] = d;
}

__global__ void edge_e_kernel(const int* __restrict__ src, const int* __restrict__ dst,
                              const float* __restrict__ a_s, const float* __restrict__ a_d,
                              float* __restrict__ ebuf, unsigned* __restrict__ menc) {
  int idx = blockIdx.x * blockDim.x + threadIdx.x;
  if (idx >= ETOT * NHEADS) return;
  int e = idx >> 2, h = idx & 3;
  int s = (e < NEDGE) ? src[e] : (e - NEDGE);
  int d = (e < NEDGE) ? dst[e] : (e - NEDGE);
  float v = a_s[s * NHEADS + h] + a_d[d * NHEADS + h];
  v = v > 0.f ? v : 0.2f * v;
  ebuf[idx] = v;
  atomicMax(&menc[d * NHEADS + h], fenc(v));
}

__global__ void edge_exp_kernel(const int* __restrict__ dst, float* __restrict__ ebuf,
                                const unsigned* __restrict__ menc, float* __restrict__ den) {
  int idx = blockIdx.x * blockDim.x + threadIdx.x;
  if (idx >= ETOT * NHEADS) return;
  int e = idx >> 2, h = idx & 3;
  int d = (e < NEDGE) ? dst[e] : (e - NEDGE);
  float ex = expf(ebuf[idx] - fdec(menc[d * NHEADS + h]));
  ebuf[idx] = ex;
  atomicAdd(&den[d * NHEADS + h], ex);
}

__global__ void edge_agg_kernel(const int* __restrict__ src, const int* __restrict__ dst,
                                const float* __restrict__ ebuf, const float* __restrict__ den,
                                const float* __restrict__ hh, float* __restrict__ agg) {
  int idx = blockIdx.x * blockDim.x + threadIdx.x;
  if (idx >= ETOT * HIDD) return;
  int e = idx >> 7, c = idx & 127, h = c >> 5;
  int s = (e < NEDGE) ? src[e] : (e - NEDGE);
  int d = (e < NEDGE) ? dst[e] : (e - NEDGE);
  float alpha = ebuf[e * NHEADS + h] / den[d * NHEADS + h];
  atomicAdd(&agg[(size_t)d * HIDD + c], alpha * hh[(size_t)s * HIDD + c]);
}

// ---------- bias + BN + (residual) + ELU ----------
__global__ void post_gat_kernel(const float* __restrict__ agg, float* __restrict__ h,
                                bf16* __restrict__ hb, const float* __restrict__ gb,
                                const float* __restrict__ bg, const float* __restrict__ bb,
                                const float* __restrict__ bm, const float* __restrict__ bv,
                                int res) {
  int idx = blockIdx.x * blockDim.x + threadIdx.x;
  if (idx >= NTOT * HIDD) return;
  int c = idx & 127;
  float v = agg[idx] + gb[c];
  v = (v - bm[c]) * rsqrtf(bv[c] + EPSV) * bg[c] + bb[c];
  if (res) v += 0.5f * h[idx];
  v = v > 0.f ? v : expm1f(v);
  h[idx] = v;
  hb[idx] = (bf16)v;
}

// ---------- SAGPool scoring ----------
__global__ void sag_dots_kernel(const float* __restrict__ h, const float* __restrict__ Wroot,
                                const float* __restrict__ Wrel, const float* __restrict__ sag_b,
                                int v, float* __restrict__ part, float* __restrict__ hwrel) {
  int n = blockIdx.x * blockDim.x + threadIdx.x;
  if (n >= NTOT) return;
  const float* hr = h + (size_t)n * HIDD;
  float r = 0.f, w = 0.f;
#pragma unroll 4
  for (int c = 0; c < HIDD; ++c) { r += hr[c] * Wroot[c]; w += hr[c] * Wrel[c]; }
  part[n] = r + sag_b[v];
  hwrel[n] = w;
}

__global__ void sag_nbr_kernel(const int* __restrict__ src, const int* __restrict__ dst,
                               const float* __restrict__ hwrel, float* __restrict__ nbr) {
  int e = blockIdx.x * blockDim.x + threadIdx.x;
  if (e >= NEDGE) return;
  atomicAdd(&nbr[dst[e]], hwrel[src[e]]);
}

__global__ void sag_fin_kernel(float* __restrict__ score, const float* __restrict__ nbr,
                               float* __restrict__ tscore) {
  int n = blockIdx.x * blockDim.x + threadIdx.x;
  if (n >= NTOT) return;
  float s = score[n] + nbr[n];
  score[n] = s;
  tscore[n] = tanhf(s);
}

// ---------- top-K(256 of 512) per graph via bitonic sort ----------
__global__ __launch_bounds__(256)
void topk_mask_kernel(const float* __restrict__ score, int* __restrict__ mask) {
  __shared__ float ss[NPG];
  __shared__ int si[NPG];
  int g = blockIdx.x, t = threadIdx.x;
  for (int r = t; r < NPG; r += 256) { ss[r] = score[g * NPG + r]; si[r] = r; }
  __syncthreads();
  for (int k = 2; k <= NPG; k <<= 1) {
    for (int j = k >> 1; j > 0; j >>= 1) {
      for (int i = t; i < NPG; i += 256) {
        int ixj = i ^ j;
        if (ixj > i) {
          float s1 = ss[i], s2 = ss[ixj];
          int i1 = si[i], i2 = si[ixj];
          bool bef = (s1 > s2) || (s1 == s2 && i1 < i2); // descending, tie: lower idx first
          bool swap = ((i & k) == 0) ? !bef : bef;
          if (swap) { ss[i] = s2; ss[ixj] = s1; si[i] = i2; si[ixj] = i1; }
        }
      }
      __syncthreads();
    }
  }
  for (int r = t; r < NPG; r += 256) mask[g * NPG + si[r]] = (r < KPOOL) ? 1 : 0;
}

__global__ void hp_kernel(const float* __restrict__ h, const float* __restrict__ tscore,
                          float* __restrict__ hp) {
  int idx = blockIdx.x * blockDim.x + threadIdx.x;
  if (idx >= NTOT * HIDD) return;
  hp[idx] = h[idx] * tscore[idx >> 7];
}

__global__ void gate_kernel(const float* __restrict__ hp, const float* __restrict__ gW,
                            const float* __restrict__ gb, int v, float* __restrict__ gate) {
  int n = blockIdx.x * blockDim.x + threadIdx.x;
  if (n >= NTOT) return;
  const float* hr = hp + (size_t)n * HIDD;
  float a = gb[v];
#pragma unroll 4
  for (int c = 0; c < HIDD; ++c) a += hr[c] * gW[c];
  gate[n] = a;
}

// ---------- masked softmax attention pooling per graph ----------
__global__ __launch_bounds__(256)
void pool_kernel(const float* __restrict__ hp, const float* __restrict__ gate,
                 const int* __restrict__ mask, float* __restrict__ views, int v) {
  __shared__ float w[NPG];
  __shared__ float red[256];
  int g = blockIdx.x, t = threadIdx.x;
  float mx = -3.0e38f;
  for (int r = t; r < NPG; r += 256) if (mask[g * NPG + r]) mx = fmaxf(mx, gate[g * NPG + r]);
  red[t] = mx; __syncthreads();
  for (int s = 128; s > 0; s >>= 1) { if (t < s) red[t] = fmaxf(red[t], red[t + s]); __syncthreads(); }
  mx = red[0]; __syncthreads();
  float sum = 0.f;
  for (int r = t; r < NPG; r += 256) {
    float e = mask[g * NPG + r] ? expf(gate[g * NPG + r] - mx) : 0.f;
    w[r] = e; sum += e;
  }
  red[t] = sum; __syncthreads();
  for (int s = 128; s > 0; s >>= 1) { if (t < s) red[t] += red[t + s]; __syncthreads(); }
  float inv = 1.f / red[0];
  __syncthreads();
  if (t < HIDD) {
    float acc = 0.f;
    for (int r = 0; r < NPG; ++r) acc += w[r] * hp[((size_t)g * NPG + r) * HIDD + t];
    views[((size_t)g * 3 + v) * HIDD + t] = acc * inv;
  }
}

// ---------- tiny 3-token MHA per graph ----------
__global__ __launch_bounds__(128)
void mha_kernel(const float* __restrict__ views, const float* __restrict__ Win,
                const float* __restrict__ bin, const float* __restrict__ Wout,
                const float* __restrict__ bout, float* __restrict__ comb) {
  __shared__ float vw[3 * HIDD];
  __shared__ float qkv[3 * 384];
  __shared__ float att[36];
  __shared__ float ob[3 * HIDD];
  int b = blockIdx.x, t = threadIdx.x;
  for (int i = t; i < 3 * HIDD; i += 128) vw[i] = views[(size_t)b * 384 + i];
  __syncthreads();
  for (int i = t; i < 3 * 384; i += 128) {
    int row = i / 384, j = i % 384;
    float acc = bin[j];
    for (int c = 0; c < HIDD; ++c) acc += vw[row * HIDD + c] * Win[(size_t)j * HIDD + c];
    qkv[i] = acc;
  }
  __syncthreads();
  if (t < 36) {
    int h = t / 9, i = (t % 9) / 3, j = t % 3;
    float acc = 0.f;
    for (int c = 0; c < HDIM; ++c)
      acc += qkv[i * 384 + h * HDIM + c] * qkv[j * 384 + 128 + h * HDIM + c];
    att[t] = acc * 0.17677669529663687f;
  }
  __syncthreads();
  if (t < 12) {
    int h = t / 3, i = t % 3;
    float* a = &att[h * 9 + i * 3];
    float m = fmaxf(a[0], fmaxf(a[1], a[2]));
    float e0 = expf(a[0] - m), e1 = expf(a[1] - m), e2 = expf(a[2] - m);
    float inv = 1.f / (e0 + e1 + e2);
    a[0] = e0 * inv; a[1] = e1 * inv; a[2] = e2 * inv;
  }
  __syncthreads();
  for (int i = t; i < 3 * HIDD; i += 128) {
    int row = i / HIDD, c = i % HIDD, h = c / HDIM;
    float acc = 0.f;
    for (int j = 0; j < 3; ++j) acc += att[h * 9 + row * 3 + j] * qkv[j * 384 + 256 + c];
    ob[i] = acc;
  }
  __syncthreads();
  for (int i = t; i < 3 * HIDD; i += 128) {
    int row = i / HIDD, j = i % HIDD;
    float acc = bout[j];
    for (int c = 0; c < HIDD; ++c) acc += ob[row * HIDD + c] * Wout[(size_t)j * HIDD + c];
    comb[(size_t)b * 384 + i] = acc;
  }
}

// ---------- classifier ----------
__global__ void fc_kernel(const float* __restrict__ in, const float* __restrict__ W,
                          const float* __restrict__ bias, float* __restrict__ out,
                          int IN, int OUT) {
  int idx = blockIdx.x * blockDim.x + threadIdx.x;
  if (idx >= BGR * OUT) return;
  int b = idx / OUT, o = idx % OUT;
  float acc = bias[o];
  for (int i = 0; i < IN; ++i) acc += in[(size_t)b * IN + i] * W[(size_t)i * OUT + o];
  out[idx] = acc;
}

__global__ void gn_relu_kernel(float* __restrict__ h, const float* __restrict__ g,
                               const float* __restrict__ bta, int C) {
  int idx = blockIdx.x * blockDim.x + threadIdx.x;
  if (idx >= BGR * 8) return;
  int b = idx / 8, grp = idx % 8, cs = C / 8;
  float* p = h + (size_t)b * C + grp * cs;
  float mu = 0.f;
  for (int i = 0; i < cs; ++i) mu += p[i];
  mu /= cs;
  float m2 = 0.f;
  for (int i = 0; i < cs; ++i) { float d = p[i] - mu; m2 += d * d; }
  float rs = rsqrtf(m2 / cs + EPSV);
  for (int i = 0; i < cs; ++i) {
    int c = grp * cs + i;
    float v = (p[i] - mu) * rs * g[c] + bta[c];
    p[i] = v > 0.f ? v : 0.f;
  }
}

// =====================================================================
extern "C" void kernel_launch(void* const* d_in, const int* in_sizes, int n_in,
                              void* d_out, int out_size, void* d_ws, size_t ws_size,
                              hipStream_t stream) {
  const float* x        = (const float*)d_in[0];
  const float* ln_g     = (const float*)d_in[1];
  const float* ln_b     = (const float*)d_in[2];
  const float* proj_W   = (const float*)d_in[3];
  const float* proj_b   = (const float*)d_in[4];
  const float* gat1_W   = (const float*)d_in[5];
  const float* gat23_W  = (const float*)d_in[6];
  const float* att_src  = (const float*)d_in[7];
  const float* att_dst  = (const float*)d_in[8];
  const float* gat_b    = (const float*)d_in[9];
  const float* bn_g     = (const float*)d_in[10];
  const float* bn_b     = (const float*)d_in[11];
  const float* bn_m     = (const float*)d_in[12];
  const float* bn_v     = (const float*)d_in[13];
  const float* sagWroot = (const float*)d_in[14];
  const float* sagWrel  = (const float*)d_in[15];
  const float* sag_b    = (const float*)d_in[16];
  const float* gate_W   = (const float*)d_in[17];
  const float* gate_b   = (const float*)d_in[18];
  const float* mha_Win  = (const float*)d_in[19];
  const float* mha_bin  = (const float*)d_in[20];
  const float* mha_Wout = (const float*)d_in[21];
  const float* mha_bout = (const float*)d_in[22];
  const float* c1_W = (const float*)d_in[23]; const float* c1_b = (const float*)d_in[24];
  const float* gn1_g = (const float*)d_in[25]; const float* gn1_b = (const float*)d_in[26];
  const float* c2_W = (const float*)d_in[27]; const float* c2_b = (const float*)d_in[28];
  const float* gn2_g = (const float*)d_in[29]; const float* gn2_b = (const float*)d_in[30];
  const float* c3_W = (const float*)d_in[31]; const float* c3_b = (const float*)d_in[32];
  const float* gn3_g = (const float*)d_in[33]; const float* gn3_b = (const float*)d_in[34];
  const float* c4_W = (const float*)d_in[35]; const float* c4_b = (const float*)d_in[36];
  const int* ei[3] = { (const int*)d_in[37], (const int*)d_in[38], (const int*)d_in[39] };

  char* base = (char*)d_ws;
  size_t off = 0;
  auto carve = [&](size_t bytes) -> void* {
    void* p = base + off;
    off += (bytes + 255) & ~(size_t)255;
    return p;
  };
  bf16*  x0b    = (bf16*)carve((size_t)NTOT * FDIM * 2);
  bf16*  pwProj = (bf16*)carve((size_t)FDIM * FDIM * 2);
  bf16*  pwG1   = (bf16*)carve((size_t)3 * FDIM * HIDD * 2);
  bf16*  pwG23  = (bf16*)carve((size_t)6 * HIDD * HIDD * 2);
  float* muB    = (float*)carve((size_t)NTOT * 4);
  float* rstdB  = (float*)carve((size_t)NTOT * 4);
  float* hB     = (float*)carve((size_t)NTOT * HIDD * 4);
  bf16*  hbB    = (bf16*)carve((size_t)NTOT * HIDD * 2);
  float* hhB    = (float*)carve((size_t)NTOT * HIDD * 4);
  float* aggB   = (float*)carve((size_t)NTOT * HIDD * 4);
  float* asB    = (float*)carve((size_t)NTOT * NHEADS * 4);
  float* adB    = (float*)carve((size_t)NTOT * NHEADS * 4);
  unsigned* mencB = (unsigned*)carve((size_t)NTOT * NHEADS * 4);
  float* denB   = (float*)carve((size_t)NTOT * NHEADS * 4);
  float* ebufB  = (float*)carve((size_t)ETOT * NHEADS * 4);
  float* scoreB = (float*)carve((size_t)NTOT * 4);
  float* tscoreB= (float*)carve((size_t)NTOT * 4);
  float* nbrB   = (float*)carve((size_t)NTOT * 4);
  float* hwrelB = (float*)carve((size_t)NTOT * 4);
  float* gateB  = (float*)carve((size_t)NTOT * 4);
  int*   maskB  = (int*)carve((size_t)NTOT * 4);
  float* viewsB = (float*)carve((size_t)BGR * 3 * HIDD * 4);
  float* combB  = (float*)carve((size_t)BGR * 3 * HIDD * 4);
  float* f1B    = (float*)carve((size_t)BGR * 512 * 4);
  float* f2B    = (float*)carve((size_t)BGR * 256 * 4);
  float* f3B    = (float*)carve((size_t)BGR * HIDD * 4);

  const int TPB = 256;
  auto cdiv = [](long long a, long long b) { return (unsigned)((a + b - 1) / b); };

  // 1) LN stats + weight repack
  row_stats_kernel<<<NTOT / 8, 256, 0, stream>>>(x, muB, rstdB);
  pack_b_kernel<<<cdiv(FDIM * FDIM, TPB), TPB, 0, stream>>>(proj_W, pwProj, FDIM, FDIM);
  for (int v = 0; v < 3; ++v)
    pack_b_kernel<<<cdiv(FDIM * HIDD, TPB), TPB, 0, stream>>>(
        gat1_W + (size_t)v * FDIM * HIDD, pwG1 + (size_t)v * FDIM * HIDD, FDIM, HIDD);
  for (int i = 0; i < 6; ++i)
    pack_b_kernel<<<cdiv(HIDD * HIDD, TPB), TPB, 0, stream>>>(
        gat23_W + (size_t)i * HIDD * HIDD, pwG23 + (size_t)i * HIDD * HIDD, HIDD, HIDD);

  // 2) LN + proj + leaky_relu -> x0 (bf16), via WMMA
  gemm_wmma_kernel<<<dim3(NTOT / 64, FDIM / 128), 128, 0, stream>>>(
      x, pwProj, nullptr, x0b, muB, rstdB, ln_g, ln_b, proj_b, FDIM, FDIM, 0);

  // 3) three views
  for (int v = 0; v < 3; ++v) {
    const int* src = ei[v];
    const int* dst = ei[v] + NEDGE;
    for (int l = 0; l < 3; ++l) {
      // hh = h @ W   (WMMA)
      if (l == 0)
        gemm_wmma_kernel<<<dim3(NTOT / 64, 1), 128, 0, stream>>>(
            x0b, pwG1 + (size_t)v * FDIM * HIDD, hhB, nullptr,
            nullptr, nullptr, nullptr, nullptr, nullptr, HIDD, FDIM, 1);
      else
        gemm_wmma_kernel<<<dim3(NTOT / 64, 1), 128, 0, stream>>>(
            hbB, pwG23 + (size_t)(v * 2 + (l - 1)) * HIDD * HIDD, hhB, nullptr,
            nullptr, nullptr, nullptr, nullptr, nullptr, HIDD, HIDD, 1);
      const float* asrc = att_src + (size_t)(v * 3 + l) * NHEADS * HDIM;
      const float* adst = att_dst + (size_t)(v * 3 + l) * NHEADS * HDIM;
      attn_coeff_kernel<<<cdiv(NTOT * NHEADS, TPB), TPB, 0, stream>>>(hhB, asrc, adst, asB, adB);
      fill_u32_kernel<<<cdiv(NTOT * NHEADS, TPB), TPB, 0, stream>>>(mencB, 0x007FFFFFu, NTOT * NHEADS);
      fill_u32_kernel<<<cdiv(NTOT * NHEADS, TPB), TPB, 0, stream>>>((unsigned*)denB, 0u, NTOT * NHEADS);
      edge_e_kernel<<<cdiv((long long)ETOT * NHEADS, TPB), TPB, 0, stream>>>(src, dst, asB, adB, ebufB, mencB);
      edge_exp_kernel<<<cdiv((long long)ETOT * NHEADS, TPB), TPB, 0, stream>>>(dst, ebufB, mencB, denB);
      fill_u32_kernel<<<cdiv(NTOT * HIDD, TPB), TPB, 0, stream>>>((unsigned*)aggB, 0u, NTOT * HIDD);
      edge_agg_kernel<<<cdiv((long long)ETOT * HIDD, TPB), TPB, 0, stream>>>(src, dst, ebufB, denB, hhB, aggB);
      const size_t po = (size_t)(v * 3 + l) * HIDD;
      post_gat_kernel<<<cdiv(NTOT * HIDD, TPB), TPB, 0, stream>>>(
          aggB, hB, hbB, gat_b + po, bn_g + po, bn_b + po, bn_m + po, bn_v + po, l > 0 ? 1 : 0);
    }
    // SAGPooling score
    sag_dots_kernel<<<cdiv(NTOT, TPB), TPB, 0, stream>>>(
        hB, sagWroot + (size_t)v * HIDD, sagWrel + (size_t)v * HIDD, sag_b, v, scoreB, hwrelB);
    fill_u32_kernel<<<cdiv(NTOT, TPB), TPB, 0, stream>>>((unsigned*)nbrB, 0u, NTOT);
    sag_nbr_kernel<<<cdiv(NEDGE, TPB), TPB, 0, stream>>>(src, dst, hwrelB, nbrB);
    sag_fin_kernel<<<cdiv(NTOT, TPB), TPB, 0, stream>>>(scoreB, nbrB, tscoreB);
    topk_mask_kernel<<<BGR, 256, 0, stream>>>(scoreB, maskB);
    hp_kernel<<<cdiv(NTOT * HIDD, TPB), TPB, 0, stream>>>(hB, tscoreB, hhB); // hp -> hhB
    gate_kernel<<<cdiv(NTOT, TPB), TPB, 0, stream>>>(hhB, gate_W + (size_t)v * HIDD, gate_b, v, gateB);
    pool_kernel<<<BGR, 256, 0, stream>>>(hhB, gateB, maskB, viewsB, v);
  }

  // 4) MHA over 3 view tokens
  mha_kernel<<<BGR, 128, 0, stream>>>(viewsB, mha_Win, mha_bin, mha_Wout, mha_bout, combB);

  // 5) classifier
  fc_kernel<<<cdiv(BGR * 512, TPB), TPB, 0, stream>>>(combB, c1_W, c1_b, f1B, 384, 512);
  gn_relu_kernel<<<cdiv(BGR * 8, TPB), TPB, 0, stream>>>(f1B, gn1_g, gn1_b, 512);
  fc_kernel<<<cdiv(BGR * 256, TPB), TPB, 0, stream>>>(f1B, c2_W, c2_b, f2B, 512, 256);
  gn_relu_kernel<<<cdiv(BGR * 8, TPB), TPB, 0, stream>>>(f2B, gn2_g, gn2_b, 256);
  fc_kernel<<<cdiv(BGR * HIDD, TPB), TPB, 0, stream>>>(f2B, c3_W, c3_b, f3B, 256, HIDD);
  gn_relu_kernel<<<cdiv(BGR * 8, TPB), TPB, 0, stream>>>(f3B, gn3_g, gn3_b, HIDD);
  fc_kernel<<<cdiv(BGR, TPB), TPB, 0, stream>>>(f3B, c4_W, c4_b, (float*)d_out, HIDD, 1);
}
